// Attention_p_2757369004155
// MI455X (gfx1250) — matrix operand order repaired and checked by reading, not compiled, not run on hardware
//
#include <hip/hip_runtime.h>
#include <hip/hip_bf16.h>

// ---------------------------------------------------------------------------
// MI455X (gfx1250, wave32) attention pipeline, all GEMMs via
// v_wmma_f32_16x16x32_bf16.  Compute-bound (~16 GFLOP over ~14MB, L2-resident).
// GEMM waves compute 16x64 tiles: one A fragment feeds 4 WMMAs (4 acc chains).
// ---------------------------------------------------------------------------

typedef __attribute__((ext_vector_type(16))) __bf16 v16bf;
typedef __attribute__((ext_vector_type(8)))  __bf16 v8bf;
typedef __attribute__((ext_vector_type(8)))  float  v8f;

#define DIMC   256
#define NHEAD  8
#define HDIM   32
#define NTOK   1728
#define NT     (NTOK / 16)        // 108 token tiles
#define NB     (NTOK / 64)        // 27  column blocks (4 tiles each)
#define NP     4
#define QSCALE 0.17677669529663687f   // 32^-0.5

// workspace layout (bytes)
#define OFF_WQKV  ((size_t)0)
#define OFF_WPROJ (OFF_WQKV  + (size_t)768 * 256 * 2)
#define OFF_XT    (OFF_WPROJ + (size_t)256 * 256 * 2)
#define OFF_QT    (OFF_XT    + (size_t)NP * NTOK * DIMC * 2)
#define OFF_KT    (OFF_QT    + (size_t)NP * NHEAD * NTOK * HDIM * 2)
#define OFF_VC    (OFF_KT    + (size_t)NP * NHEAD * NTOK * HDIM * 2)
#define OFF_CTX   (OFF_VC    + (size_t)NP * NHEAD * HDIM * NTOK * 2)

__device__ inline v8f wmma_bf16(v16bf a, v16bf b, v8f c) {
  return __builtin_amdgcn_wmma_f32_16x16x32_bf16(
      /*neg_a=*/false, a, /*neg_b=*/false, b,
      /*c_mod=*/(short)0, c, /*reuse_a=*/false, /*reuse_b=*/false);
}

// A-matrix 16x32 bf16 fragment (ISA 7.12.2): lane&15 = row M,
// per-lane K chunks at koff=(lane>>4)*8: [koff..koff+7] and [koff+16..koff+23].
__device__ inline v16bf load_a_frag(const __bf16* base, int row_stride,
                                    int rowbase, int kbase, int lane) {
  int row  = rowbase + (lane & 15);
  int koff = (lane >> 4) * 8;
  const __bf16* rp = base + (size_t)row * row_stride + kbase + koff;
  v8bf c0 = *(const v8bf*)(rp);
  v8bf c1 = *(const v8bf*)(rp + 16);
  return __builtin_shufflevector(c0, c1, 0, 1, 2, 3, 4, 5, 6, 7,
                                 8, 9, 10, 11, 12, 13, 14, 15);
}

// B-matrix 32x16 bf16 fragment: lane&15 = col N, per-lane K chunks at
// koff=(lane>>4)*8: [koff..koff+7] and [koff+16..koff+23].  `base` is indexed [col][k] (k contiguous).
__device__ inline v16bf load_b_frag(const __bf16* base, int col_stride,
                                    int colbase, int kbase, int lane) {
  int col = colbase + (lane & 15);
  const __bf16* cp = base + (size_t)col * col_stride + kbase + (lane >> 4) * 8;
  v8bf c0 = *(const v8bf*)(cp);
  v8bf c1 = *(const v8bf*)(cp + 16);
  return __builtin_shufflevector(c0, c1, 0, 1, 2, 3, 4, 5, 6, 7,
                                 8, 9, 10, 11, 12, 13, 14, 15);
}

// ---------------------------------------------------------------------------
// conversion kernels
// ---------------------------------------------------------------------------
__global__ void conv_weights_kernel(const float* __restrict__ wqkv,
                                    const float* __restrict__ wproj,
                                    __bf16* __restrict__ wq_b,
                                    __bf16* __restrict__ wp_b) {
  int i = blockIdx.x * blockDim.x + threadIdx.x;
  if (i < 768 * 256) wq_b[i] = (__bf16)wqkv[i];
  if (i < 256 * 256) wp_b[i] = (__bf16)wproj[i];
}

// x [P][C][N] f32 -> xT [P][N][C] bf16 (token-major for B fragments)
__global__ void conv_x_kernel(const float* __restrict__ x,
                              __bf16* __restrict__ xT) {
  int i   = blockIdx.x * blockDim.x + threadIdx.x;  // exact P*C*N threads
  int p   = i / (DIMC * NTOK);
  int rem = i % (DIMC * NTOK);
  int c   = rem / NTOK;
  int n   = rem % NTOK;
  xT[((size_t)p * NTOK + n) * DIMC + c] = (__bf16)x[i];
}

// ---------------------------------------------------------------------------
// QKV GEMM: [768x256] x [256x1728] per p.  One 16x64 tile per wave:
// A fragment reused by 4 WMMAs per K step, 4 independent accumulators.
// q,k stored token-major [p][h][n][hd]; v channel-major [p][h][hd][n].
// ---------------------------------------------------------------------------
__global__ void __launch_bounds__(256)
qkv_gemm_kernel(const __bf16* __restrict__ wqkv, const __bf16* __restrict__ xT,
                __bf16* __restrict__ qT, __bf16* __restrict__ kT,
                __bf16* __restrict__ vC) {
  int wave = blockIdx.x * 8 + (threadIdx.x >> 5);
  int lane = threadIdx.x & 31;
  int p    = wave / (48 * NB);
  int rem  = wave % (48 * NB);
  int mt   = rem / NB;          // 16-row tile of output channels
  int nb   = rem % NB;          // 64-col block of tokens
  const __bf16* xTp = xT + (size_t)p * NTOK * DIMC;

  v8f acc[4] = {{}, {}, {}, {}};
#pragma unroll
  for (int kb = 0; kb < DIMC; kb += 32) {
    v16bf a = load_a_frag(wqkv, DIMC, mt * 16, kb, lane);
#pragma unroll
    for (int j = 0; j < 4; ++j) {
      v16bf b = load_b_frag(xTp, DIMC, nb * 64 + j * 16, kb, lane);
      acc[j] = wmma_bf16(a, b, acc[j]);
    }
  }

  int obase = mt * 16 + 8 * (lane >> 4);   // output-channel base for this lane
  int sel   = obase >> 8;                  // 0=q 1=k 2=v (constant per tile)
  int oc    = obase & 255;
  int h     = oc >> 5;
#pragma unroll
  for (int j = 0; j < 4; ++j) {
    int n = nb * 64 + j * 16 + (lane & 15);
#pragma unroll
    for (int i = 0; i < 8; ++i) {
      int   hd = (oc & 31) + i;
      float v  = acc[j][i];
      if (sel == 0) {
        qT[(((size_t)p * NHEAD + h) * NTOK + n) * HDIM + hd] =
            (__bf16)(v * QSCALE);
      } else if (sel == 1) {
        kT[(((size_t)p * NHEAD + h) * NTOK + n) * HDIM + hd] = (__bf16)v;
      } else {
        vC[(((size_t)p * NHEAD + h) * HDIM + hd) * NTOK + n] = (__bf16)v;
      }
    }
  }
}

// ---------------------------------------------------------------------------
// Attention: per wave = one (p, h, 16-query tile).  Computes S^T = K x Q^T so
// the softmax axis is the register axis of the C tile: row max/sum = per-lane
// reduce over 8 VGPRs + one shfl_xor(16).  Two-pass softmax; exp'd scores are
// repacked in-register as the A fragment of the PV WMMA (no LDS, no N x N).
// ---------------------------------------------------------------------------
__global__ void __launch_bounds__(256)
attn_kernel(const __bf16* __restrict__ qT, const __bf16* __restrict__ kT,
            const __bf16* __restrict__ vC, __bf16* __restrict__ ctxT) {
  int wave = blockIdx.x * 8 + (threadIdx.x >> 5);
  int lane = threadIdx.x & 31;
  int p    = wave / (NHEAD * NT);
  int rem  = wave % (NHEAD * NT);
  int h    = rem / NT;
  int mt   = rem % NT;

  const __bf16* q_ph = qT + ((size_t)p * NHEAD + h) * NTOK * HDIM;
  const __bf16* k_ph = kT + ((size_t)p * NHEAD + h) * NTOK * HDIM;
  const __bf16* v_ph = vC + ((size_t)p * NHEAD + h) * HDIM * NTOK;

  // Q as the B operand (col = query token m, K = head dim), loaded once.
  v16bf bq = load_b_frag(q_ph, HDIM, mt * 16, 0, lane);

  // ---- pass 1: per-query max over all keys --------------------------------
  float rmax = -3.0e38f;
  for (int t = 0; t < NT; ++t) {
    __builtin_prefetch(k_ph + (size_t)(t + 4 < NT ? t + 4 : t) * 16 * HDIM, 0, 1);
    v16bf ak = load_a_frag(k_ph, HDIM, t * 16, 0, lane);
    v8f   z  = {};
    v8f   st = wmma_bf16(ak, bq, z);   // st[n, m]: lane=col m, regs=rows n
#pragma unroll
    for (int i = 0; i < 8; ++i) rmax = fmaxf(rmax, st[i]);
  }
  rmax = fmaxf(rmax, __shfl_xor(rmax, 16, 32));

  // ---- pass 2: exp, sum, PV accumulate ------------------------------------
  float rsum = 0.f;
  v8f acc0 = {}, acc1 = {};
  for (int t = 0; t < NT; t += 2) {
    __builtin_prefetch(v_ph + (size_t)(t + 2 < NT ? t + 2 : t) * 16, 0, 1);
    v16bf ak0 = load_a_frag(k_ph, HDIM, t * 16, 0, lane);
    v16bf ak1 = load_a_frag(k_ph, HDIM, (t + 1) * 16, 0, lane);
    v8f   z   = {};
    v8f   s0  = wmma_bf16(ak0, bq, z);
    v8f   s1  = wmma_bf16(ak1, bq, z);
    v16bf ap;  // A fragment of PV: K = 32 keys (tiles t, t+1), M = 16 queries
#pragma unroll
    for (int i = 0; i < 8; ++i) {
      float e0 = __expf(s0[i] - rmax);
      float e1 = __expf(s1[i] - rmax);
      rsum += e0 + e1;
      ap[i]     = (__bf16)e0;   // regs -> K 0..7 / 8..15 (lane-half)
      ap[i + 8] = (__bf16)e1;   // regs -> K 16..23 / 24..31
    }
    v16bf bv0 = load_b_frag(v_ph, NTOK, 0,  t * 16, lane);  // channels 0..15
    v16bf bv1 = load_b_frag(v_ph, NTOK, 16, t * 16, lane);  // channels 16..31
    acc0 = wmma_bf16(ap, bv0, acc0);
    acc1 = wmma_bf16(ap, bv1, acc1);
  }
  rsum += __shfl_xor(rsum, 16, 32);
  float inv = 1.0f / rsum;

  // acc tiles: lane = channel col, regs = query rows -> fetch 1/sum per row.
#pragma unroll
  for (int i = 0; i < 8; ++i) {
    int   mloc = i + 8 * (lane >> 4);
    float f    = __shfl(inv, mloc, 32);
    int   m    = mt * 16 + mloc;
    int   c    = lane & 15;
    size_t row = ((size_t)p * NTOK + m) * DIMC + (size_t)h * HDIM;
    ctxT[row + c]      = (__bf16)(acc0[i] * f);
    ctxT[row + 16 + c] = (__bf16)(acc1[i] * f);
  }
}

// ---------------------------------------------------------------------------
// Output projection: [256x256] x ctx[256x1728] + bias -> f32 output [p][o][n]
// 16x64 tile per wave, A fragment reused by 4 WMMAs per K step.
// ---------------------------------------------------------------------------
__global__ void __launch_bounds__(256)
proj_gemm_kernel(const __bf16* __restrict__ wproj,
                 const __bf16* __restrict__ ctxT,
                 const float* __restrict__ bias, float* __restrict__ out) {
  int wave = blockIdx.x * 8 + (threadIdx.x >> 5);
  int lane = threadIdx.x & 31;
  int p    = wave / (16 * NB);
  int rem  = wave % (16 * NB);
  int mt   = rem / NB;
  int nb   = rem % NB;
  const __bf16* cp = ctxT + (size_t)p * NTOK * DIMC;

  v8f acc[4] = {{}, {}, {}, {}};
#pragma unroll
  for (int kb = 0; kb < DIMC; kb += 32) {
    v16bf a = load_a_frag(wproj, DIMC, mt * 16, kb, lane);
#pragma unroll
    for (int j = 0; j < 4; ++j) {
      v16bf b = load_b_frag(cp, DIMC, nb * 64 + j * 16, kb, lane);
      acc[j] = wmma_bf16(a, b, acc[j]);
    }
  }

  int ob = mt * 16 + 8 * (lane >> 4);
#pragma unroll
  for (int j = 0; j < 4; ++j) {
    int n = nb * 64 + j * 16 + (lane & 15);
#pragma unroll
    for (int i = 0; i < 8; ++i) {
      int o = ob + i;
      out[((size_t)p * DIMC + o) * NTOK + n] = acc[j][i] + bias[o];
    }
  }
}

// ---------------------------------------------------------------------------
extern "C" void kernel_launch(void* const* d_in, const int* in_sizes, int n_in,
                              void* d_out, int out_size, void* d_ws,
                              size_t ws_size, hipStream_t stream) {
  (void)in_sizes; (void)n_in; (void)out_size; (void)ws_size;
  const float* x      = (const float*)d_in[0];
  const float* w_qkv  = (const float*)d_in[1];
  const float* w_proj = (const float*)d_in[2];
  const float* b_proj = (const float*)d_in[3];
  float*       out    = (float*)d_out;

  char* ws = (char*)d_ws;
  __bf16* wq_b = (__bf16*)(ws + OFF_WQKV);
  __bf16* wp_b = (__bf16*)(ws + OFF_WPROJ);
  __bf16* xT   = (__bf16*)(ws + OFF_XT);
  __bf16* qT   = (__bf16*)(ws + OFF_QT);
  __bf16* kT   = (__bf16*)(ws + OFF_KT);
  __bf16* vC   = (__bf16*)(ws + OFF_VC);
  __bf16* ctxT = (__bf16*)(ws + OFF_CTX);

  // 1) convert weights + x to bf16 (x transposed token-major)
  conv_weights_kernel<<<(768 * 256 + 255) / 256, 256, 0, stream>>>(
      w_qkv, w_proj, wq_b, wp_b);
  conv_x_kernel<<<(NP * DIMC * NTOK) / 256, 256, 0, stream>>>(x, xT);

  // 2) QKV projection: 4*48*27 wave-tiles (16x64 each), 8 waves per block
  qkv_gemm_kernel<<<(NP * 48 * NB) / 8, 256, 0, stream>>>(wq_b, xT, qT, kT, vC);

  // 3) attention: 4*8*108 wave-jobs
  attn_kernel<<<(NP * NHEAD * NT) / 8, 256, 0, stream>>>(qT, kT, vC, ctxT);

  // 4) output projection + bias: 4*16*27 wave-tiles (16x64 each)
  proj_gemm_kernel<<<(NP * 16 * NB) / 8, 256, 0, stream>>>(wp_b, ctxT, b_proj,
                                                           out);
}
